// VQFPEnhancer_PCNN_27779848470825
// MI455X (gfx1250) — matrix operand order repaired and checked by reading, not compiled, not run on hardware
//
#include <hip/hip_runtime.h>
#include <hip/hip_bf16.h>
#include <stdint.h>

typedef __attribute__((ext_vector_type(16))) _Float16 v16h;
typedef __attribute__((ext_vector_type(8)))  _Float16 v8h;
typedef __attribute__((ext_vector_type(8)))  float    v8f;

#define B_    4
#define ZC_   64
#define E_    16
#define NE_   16384
#define PC_   64
#define FR_   32
#define HW_   (FR_*FR_)      // 1024
#define NPIX  (B_*HW_)       // 4096

// Branchless order-preserving float->uint mapping (monotone: a<b  <=>  ord(a)<ord(b))
__device__ __forceinline__ uint32_t f32_orderable(float f) {
  uint32_t u = __float_as_uint(f);
  return u ^ (uint32_t)(((int32_t)u >> 31) | 0x80000000);
}

__device__ __forceinline__ unsigned long long umin64(unsigned long long a, unsigned long long b) {
  return a < b ? a : b;
}
__device__ __forceinline__ unsigned long long umax64(unsigned long long a, unsigned long long b) {
  return a > b ? a : b;
}

__device__ __forceinline__ v16h cat8(v8h lo, v8h hi) {
  return __builtin_shufflevector(lo, hi, 0,1,2,3,4,5,6,7,8,9,10,11,12,13,14,15);
}

// ---------------------------------------------------------------- init keys
__global__ void k_init(unsigned long long* keys_vq, unsigned long long* keys_pcn) {
  int i = blockIdx.x * blockDim.x + threadIdx.x;
  if (i < NPIX) { keys_vq[i] = ~0ull; keys_pcn[i] = 0ull; }
}

// ------------------------------------------- codebook -> f16 (K padded 32) + norms
__global__ void k_cb(const float* __restrict__ cb, _Float16* __restrict__ cbh,
                     float* __restrict__ cnorm) {
  int n = blockIdx.x * blockDim.x + threadIdx.x;
  if (n >= NE_) return;
  float s = 0.f;
  #pragma unroll
  for (int e = 0; e < E_; ++e) {
    float v = cb[n * E_ + e];
    s += v * v;
    cbh[n * 32 + e] = (_Float16)v;
  }
  #pragma unroll
  for (int e = E_; e < 32; ++e) cbh[n * 32 + e] = (_Float16)0.f;
  cnorm[n] = s;
}

// ------------------------------------------- quant_conv 1x1 -> z_f16 [p][32] padded
__global__ void k_quant(const float* __restrict__ z_t, const float* __restrict__ qw,
                        const float* __restrict__ qb, _Float16* __restrict__ zh) {
  int t = blockIdx.x * blockDim.x + threadIdx.x;       // p*32 + slot
  if (t >= NPIX * 32) return;
  int p = t >> 5, slot = t & 31;
  if (slot >= E_) { zh[t] = (_Float16)0.f; return; }
  int b = p / HW_, yx = p % HW_;
  const float* src  = z_t + (size_t)b * ZC_ * HW_ + yx;
  const float* wrow = qw + slot * ZC_;
  float acc = qb[slot];
  #pragma unroll 8
  for (int c = 0; c < ZC_; ++c) acc += wrow[c] * src[c * HW_];
  zh[t] = (_Float16)acc;
}

// ------------------------------------------- lr_quant_conv 1x1 -> z_l (f32 NCHW, output)
__global__ void k_lrq(const float* __restrict__ zpre, const float* __restrict__ w,
                      const float* __restrict__ bb, float* __restrict__ z_l) {
  int t = blockIdx.x * blockDim.x + threadIdx.x;       // p*16 + e
  if (t >= NPIX * E_) return;
  int p = t >> 4, e = t & 15;
  int b = p / HW_, yx = p % HW_;
  const float* src  = zpre + (size_t)b * ZC_ * HW_ + yx;
  const float* wrow = w + e * ZC_;
  float acc = bb[e];
  #pragma unroll 8
  for (int c = 0; c < ZC_; ++c) acc += wrow[c] * src[c * HW_];
  z_l[(b * E_ + e) * HW_ + yx] = acc;
}

// ------------------------------------------- PixelCNN body conv7x7 + relu -> h_f16 [p][64]
__global__ void k_pcn1(const float* __restrict__ z_l, const float* __restrict__ w1,
                       const float* __restrict__ b1, _Float16* __restrict__ hh) {
  int t = blockIdx.x * blockDim.x + threadIdx.x;       // p*64 + oc
  if (t >= NPIX * PC_) return;
  int p = t >> 6, oc = t & 63;
  int b = p / HW_, yx = p % HW_, y = yx >> 5, x = yx & 31;
  float acc = b1[oc];
  for (int ic = 0; ic < E_; ++ic) {
    const float* in = z_l + (b * E_ + ic) * HW_;
    const float* wv = w1 + (oc * E_ + ic) * 49;
    #pragma unroll
    for (int ky = 0; ky < 7; ++ky) {
      int yy = y + ky - 3;
      if ((unsigned)yy >= (unsigned)FR_) continue;
      #pragma unroll
      for (int kx = 0; kx < 7; ++kx) {
        int xx = x + kx - 3;
        if ((unsigned)xx >= (unsigned)FR_) continue;
        acc += wv[ky * 7 + kx] * in[yy * FR_ + xx];
      }
    }
  }
  hh[t] = (_Float16)fmaxf(acc, 0.f);
}

// ------------------------------------------- pcn_w2 f32 -> f16
__global__ void k_w2(const float* __restrict__ w2, _Float16* __restrict__ w2h) {
  int i = blockIdx.x * blockDim.x + threadIdx.x;
  if (i < NE_ * PC_) w2h[i] = (_Float16)w2[i];
}

// ------------------------------------------- VQ distance GEMM (WMMA) + fused argmin
// dist = |c|^2 - 2 z.c (|z|^2 constant per row). A = z_f16 [4096x32], B = cb_f16 [16384x32].
// Best candidate tracked as packed u64 key (orderable(score)<<32 | n): branchless min.
__global__ void k_vq_gemm(const _Float16* __restrict__ zh, const _Float16* __restrict__ cbh,
                          const float* __restrict__ cnorm, unsigned long long* __restrict__ keys) {
  int wave = blockIdx.x * (blockDim.x >> 5) + (threadIdx.x >> 5);
  int lane = threadIdx.x & 31;
  const int NBLK = NE_ / 64;                    // 256 column blocks of 64
  int mt = wave / NBLK, nb = wave % NBLK;
  int row0 = mt * 16, col0 = nb * 64;
  int lo = lane & 15, hi = lane >> 4;
  int mrow = row0 + lo;
  int kb = 8 * hi;
  // A fragment (16-bit A 16x32 layout: lanes 0-15 K{0..7,16..23}, lanes 16-31 K{8..15,24..31})
  v8h alo = *(const v8h*)(zh + mrow * 32 + kb);
  v8h ahi = *(const v8h*)(zh + mrow * 32 + 16 + kb);
  v16h a = cat8(alo, ahi);
  int kb2 = 16 * hi;

  unsigned long long best[8];
  #pragma unroll
  for (int r = 0; r < 8; ++r) best[r] = ~0ull;

  #pragma unroll
  for (int nt = 0; nt < 4; ++nt) {
    int n = col0 + nt * 16 + lo;
    v16h bf = cat8(*(const v8h*)(cbh + n * 32 + kb), *(const v8h*)(cbh + n * 32 + 16 + kb));   // B 32x16: same K order as A
    v8f c = {};
    c = __builtin_amdgcn_wmma_f32_16x16x32_f16(false, a, false, bf, (short)0, c, false, false);
    float cn = cnorm[n];
    #pragma unroll
    for (int r = 0; r < 8; ++r) {
      float score = fmaf(-2.0f, c[r], cn);
      unsigned long long key =
          ((unsigned long long)f32_orderable(score) << 32) | (uint32_t)n;
      best[r] = umin64(best[r], key);               // min score, tie -> lowest index
    }
  }
  // reduce across the 16 column lanes of each half-wave (branchless u64 min)
  #pragma unroll
  for (int off = 1; off < 16; off <<= 1) {
    #pragma unroll
    for (int r = 0; r < 8; ++r) {
      unsigned long long o = __shfl_xor(best[r], off, 32);
      best[r] = umin64(best[r], o);
    }
  }
  if (lo == 0) {
    #pragma unroll
    for (int r = 0; r < 8; ++r) {
      int row = row0 + r + 8 * hi;
      atomicMin(&keys[row], best[r]);
    }
  }
}

// ------------------------------------------- target_index + z_hq gather
__global__ void k_vq_out(const unsigned long long* __restrict__ keys,
                         const float* __restrict__ cb,
                         float* __restrict__ target_index, float* __restrict__ z_hq) {
  int t = blockIdx.x * blockDim.x + threadIdx.x;   // p*16 + e
  if (t >= NPIX * E_) return;
  int p = t >> 4, e = t & 15;
  int idx = (int)(uint32_t)(keys[p] & 0xFFFFFFFFull);
  if (e == 0) target_index[p] = (float)idx;
  int b = p / HW_, yx = p % HW_;
  z_hq[(b * E_ + e) * HW_ + yx] = cb[idx * E_ + e];
}

// ------------------------------------------- logits GEMM (WMMA) + fused argmax
// logits[row][n] = h[row][:] . w2[n][:] + b2[n];  row = b*1024 + y*32 + x
// Best tracked as packed u64 key (orderable(val)<<32 | ~n): branchless max, tie -> lowest n.
__global__ void k_pcn_gemm(const _Float16* __restrict__ hh, const _Float16* __restrict__ w2h,
                           const float* __restrict__ b2, float* __restrict__ logits,
                           unsigned long long* __restrict__ keys) {
  int wave = blockIdx.x * (blockDim.x >> 5) + (threadIdx.x >> 5);
  int lane = threadIdx.x & 31;
  const int NBLK = NE_ / 64;
  int mt = wave / NBLK, nb = wave % NBLK;
  int row0 = mt * 16, col0 = nb * 64;
  int lo = lane & 15, hi = lane >> 4;
  int mrow = row0 + lo;
  int kb = 8 * hi;
  const _Float16* arow = hh + mrow * 64;
  v16h a0 = cat8(*(const v8h*)(arow + kb),      *(const v8h*)(arow + 16 + kb));   // K 0..31
  v16h a1 = cat8(*(const v8h*)(arow + 32 + kb), *(const v8h*)(arow + 48 + kb));   // K 32..63
  int kb2 = 16 * hi;

  unsigned long long best[8];
  #pragma unroll
  for (int r = 0; r < 8; ++r) best[r] = 0ull;

  #pragma unroll
  for (int nt = 0; nt < 4; ++nt) {
    int n = col0 + nt * 16 + lo;
    const _Float16* brow = w2h + n * 64;
    v16h b0 = cat8(*(const v8h*)(brow + kb),      *(const v8h*)(brow + 16 + kb));
    v16h b1 = cat8(*(const v8h*)(brow + 32 + kb), *(const v8h*)(brow + 48 + kb));
    v8f c = {};
    c = __builtin_amdgcn_wmma_f32_16x16x32_f16(false, a0, false, b0, (short)0, c, false, false);
    c = __builtin_amdgcn_wmma_f32_16x16x32_f16(false, a1, false, b1, (short)0, c, false, false);
    float bias = b2[n];
    #pragma unroll
    for (int r = 0; r < 8; ++r) {
      int row = row0 + r + 8 * hi;
      float val = c[r] + bias;
      logits[(size_t)row * NE_ + n] = val;
      unsigned long long key =
          ((unsigned long long)f32_orderable(val) << 32) | (uint32_t)(~(uint32_t)n);
      best[r] = umax64(best[r], key);               // max logit, tie -> lowest index
    }
  }
  #pragma unroll
  for (int off = 1; off < 16; off <<= 1) {
    #pragma unroll
    for (int r = 0; r < 8; ++r) {
      unsigned long long o = __shfl_xor(best[r], off, 32);
      best[r] = umax64(best[r], o);
    }
  }
  if (lo == 0) {
    #pragma unroll
    for (int r = 0; r < 8; ++r) {
      int row = row0 + r + 8 * hi;
      atomicMax(&keys[row], best[r]);
    }
  }
}

// ------------------------------------------- z_lq = post_quant_conv(codebook[top_index])
__global__ void k_zlq(const unsigned long long* __restrict__ keys, const float* __restrict__ cb,
                      const float* __restrict__ pqw, const float* __restrict__ pqb,
                      float* __restrict__ zlq) {
  int t = blockIdx.x * blockDim.x + threadIdx.x;   // p*64 + oc
  if (t >= NPIX * ZC_) return;
  int p = t >> 6, oc = t & 63;
  int idx = (int)(~(uint32_t)(keys[p] & 0xFFFFFFFFull));
  const float* wrow = pqw + oc * E_;
  const float* crow = cb + idx * E_;
  float acc = pqb[oc];
  #pragma unroll
  for (int e = 0; e < E_; ++e) acc += wrow[e] * crow[e];
  int b = p / HW_, yx = p % HW_;
  zlq[(b * ZC_ + oc) * HW_ + yx] = acc;
}

// ------------------------------------------- d1 = relu(conv3x3(z_lq)) at 32x32
__global__ void k_dec1(const float* __restrict__ zlq, const float* __restrict__ w,
                       const float* __restrict__ bias, float* __restrict__ d1) {
  int t = blockIdx.x * blockDim.x + threadIdx.x;   // (b*64+oc)*1024 + yx
  if (t >= B_ * ZC_ * HW_) return;
  int yx = t & 1023, boc = t >> 10, oc = boc & 63, b = boc >> 6;
  int y = yx >> 5, x = yx & 31;
  float acc = bias[oc];
  for (int ic = 0; ic < ZC_; ++ic) {
    const float* in = zlq + (b * ZC_ + ic) * HW_;
    const float* wv = w + (oc * ZC_ + ic) * 9;
    #pragma unroll
    for (int ky = 0; ky < 3; ++ky) {
      int yy = y + ky - 1;
      if ((unsigned)yy >= (unsigned)FR_) continue;
      #pragma unroll
      for (int kx = 0; kx < 3; ++kx) {
        int xx = x + kx - 1;
        if ((unsigned)xx >= (unsigned)FR_) continue;
        acc += wv[ky * 3 + kx] * in[yy * FR_ + xx];
      }
    }
  }
  d1[t] = fmaxf(acc, 0.f);
}

// ------------------------------------------- d2 = up2(d1) + w*conv3x3(fuse_feat) at 64x64
__global__ void k_fuse(const float* __restrict__ d1, const float* __restrict__ ff,
                       const float* __restrict__ fw, const float* __restrict__ fb,
                       const int* __restrict__ wscale, float* __restrict__ d2) {
  int t = blockIdx.x * blockDim.x + threadIdx.x;   // (b*64+oc)*4096 + YX
  if (t >= B_ * ZC_ * 4096) return;
  int YX = t & 4095, boc = t >> 12, oc = boc & 63, b = boc >> 6;
  int Y = YX >> 6, X = YX & 63;
  float acc = fb[oc];
  for (int ic = 0; ic < ZC_; ++ic) {
    const float* in = ff + (size_t)(b * ZC_ + ic) * 4096;
    const float* wv = fw + (oc * ZC_ + ic) * 9;
    #pragma unroll
    for (int ky = 0; ky < 3; ++ky) {
      int yy = Y + ky - 1;
      if ((unsigned)yy >= 64u) continue;
      #pragma unroll
      for (int kx = 0; kx < 3; ++kx) {
        int xx = X + kx - 1;
        if ((unsigned)xx >= 64u) continue;
        acc += wv[ky * 3 + kx] * in[yy * 64 + xx];
      }
    }
  }
  float ws = (float)wscale[0];
  d2[t] = d1[(b * ZC_ + oc) * HW_ + (Y >> 1) * FR_ + (X >> 1)] + ws * acc;
}

// ------------------------------------------- dec = conv3x3(up2(relu(d2))) at 128x128, 1 ch
__global__ void k_dec2(const float* __restrict__ d2, const float* __restrict__ w,
                       const float* __restrict__ bias, float* __restrict__ dec) {
  int t = blockIdx.x * blockDim.x + threadIdx.x;   // b*16384 + i*128 + j
  if (t >= B_ * 128 * 128) return;
  int ij = t & 16383, b = t >> 14, i = ij >> 7, j = ij & 127;
  float acc = bias[0];
  for (int c = 0; c < ZC_; ++c) {
    const float* in = d2 + (size_t)(b * ZC_ + c) * 4096;
    const float* wv = w + c * 9;
    #pragma unroll
    for (int di = 0; di < 3; ++di) {
      int p = i + di - 1;
      if ((unsigned)p >= 128u) continue;
      int P = p >> 1;
      #pragma unroll
      for (int dj = 0; dj < 3; ++dj) {
        int q = j + dj - 1;
        if ((unsigned)q >= 128u) continue;
        int Q = q >> 1;
        acc += wv[di * 3 + dj] * fmaxf(in[P * 64 + Q], 0.f);
      }
    }
  }
  dec[t] = acc;
}

// ================================================================= launch
extern "C" void kernel_launch(void* const* d_in, const int* in_sizes, int n_in,
                              void* d_out, int out_size, void* d_ws, size_t ws_size,
                              hipStream_t stream) {
  const float* z_t     = (const float*)d_in[0];
  const float* z_l_pre = (const float*)d_in[1];
  const float* ff      = (const float*)d_in[2];
  const int*   wsc     = (const int*)  d_in[3];
  const float* quant_w = (const float*)d_in[4];
  const float* quant_b = (const float*)d_in[5];
  const float* cb      = (const float*)d_in[6];
  const float* lrq_w   = (const float*)d_in[7];
  const float* lrq_b   = (const float*)d_in[8];
  const float* pcn_w1  = (const float*)d_in[9];
  const float* pcn_b1  = (const float*)d_in[10];
  const float* pcn_w2  = (const float*)d_in[11];
  const float* pcn_b2  = (const float*)d_in[12];
  const float* pq_w    = (const float*)d_in[13];
  const float* pq_b    = (const float*)d_in[14];
  const float* dec_w1  = (const float*)d_in[15];
  const float* dec_b1  = (const float*)d_in[16];
  const float* fuse_w  = (const float*)d_in[17];
  const float* fuse_b  = (const float*)d_in[18];
  const float* dec_w2  = (const float*)d_in[19];
  const float* dec_b2  = (const float*)d_in[20];

  // Output layout (flat concat, return order): logits, z_l, target_index, z_hq, dec
  float* out      = (float*)d_out;
  float* logits   = out;                                    // 4*1024*16384
  float* z_l_out  = logits + (size_t)NPIX * NE_;            // 4*16*32*32
  float* tidx_out = z_l_out + NPIX * E_;                    // 4096
  float* zhq_out  = tidx_out + NPIX;                        // 4*16*32*32
  float* dec_out  = zhq_out + NPIX * E_;                    // 4*1*128*128

  // Workspace layout (bytes)
  char* ws = (char*)d_ws;
  _Float16* cbh   = (_Float16*)(ws + 0);                    // 1,048,576
  float*    cnorm = (float*)   (ws + 1048576);              //    65,536
  _Float16* zh    = (_Float16*)(ws + 1114112);              //   262,144
  unsigned long long* keys_vq  = (unsigned long long*)(ws + 1376256);  // 32,768
  unsigned long long* keys_pcn = (unsigned long long*)(ws + 1409024);  // 32,768
  _Float16* hh    = (_Float16*)(ws + 1441792);              //   524,288
  _Float16* w2h   = (_Float16*)(ws + 1966080);              // 2,097,152
  float*    zlq   = (float*)   (ws + 4063232);              // 1,048,576
  float*    d1    = (float*)   (ws + 5111808);              // 1,048,576
  float*    d2    = (float*)   (ws + 6160384);              // 4,194,304

  const int T = 256;
  k_init <<<(NPIX + T - 1) / T, T, 0, stream>>>(keys_vq, keys_pcn);
  k_cb   <<<(NE_ + T - 1) / T, T, 0, stream>>>(cb, cbh, cnorm);
  k_quant<<<(NPIX * 32) / T, T, 0, stream>>>(z_t, quant_w, quant_b, zh);
  k_lrq  <<<(NPIX * E_) / T, T, 0, stream>>>(z_l_pre, lrq_w, lrq_b, z_l_out);
  k_pcn1 <<<(NPIX * PC_) / T, T, 0, stream>>>(z_l_out, pcn_w1, pcn_b1, hh);
  k_w2   <<<(NE_ * PC_) / T, T, 0, stream>>>(pcn_w2, w2h);

  // 65536 wave-jobs each, 8 waves per 256-thread block -> 8192 blocks
  k_vq_gemm <<<8192, T, 0, stream>>>(zh, cbh, cnorm, keys_vq);
  k_vq_out  <<<(NPIX * E_) / T, T, 0, stream>>>(keys_vq, cb, tidx_out, zhq_out);

  k_pcn_gemm<<<8192, T, 0, stream>>>(hh, w2h, pcn_b2, logits, keys_pcn);

  k_zlq  <<<(NPIX * ZC_) / T, T, 0, stream>>>(keys_pcn, cb, pq_w, pq_b, zlq);
  k_dec1 <<<(B_ * ZC_ * HW_) / T, T, 0, stream>>>(zlq, dec_w1, dec_b1, d1);
  k_fuse <<<(B_ * ZC_ * 4096) / T, T, 0, stream>>>(d1, ff, fuse_w, fuse_b, wsc, d2);
  k_dec2 <<<(B_ * 128 * 128) / T, T, 0, stream>>>(d2, dec_w2, dec_b2, dec_out);
}
